// CASCOAttention_18485539242617
// MI455X (gfx1250) — matrix operand order repaired and checked by reading, not compiled, not run on hardware
//
#include <hip/hip_runtime.h>
#include <hip/hip_bf16.h>

// ---------------------------------------------------------------------------
// CASCO attention for MI455X (gfx1250): bf16 WMMA, flash-style streaming
// softmax with lane-local online stats, async global->LDS tile staging
// (double-buffered), V stored per-head transposed so tiles need no shuffle.
// ---------------------------------------------------------------------------

typedef unsigned short u16;
typedef __attribute__((ext_vector_type(16))) __bf16 v16bf;
typedef __attribute__((ext_vector_type(8)))  float  v8f;

#define BATCH 4
#define SEQ   2048
#define EMB   768
#define NH    12
#define HD    64

static __device__ __forceinline__ u16 f2bf(float f) {
  unsigned u = __float_as_uint(f);
  unsigned r = u + 0x7FFFu + ((u >> 16) & 1u);   // round-to-nearest-even
  return (u16)(r >> 16);
}

static __device__ __forceinline__ v8f wmma_bf16(v16bf a, v16bf b, v8f c) {
  return __builtin_amdgcn_wmma_f32_16x16x32_bf16(false, a, false, b, (short)0, c,
                                                 false, false);
}

union Frag16 { v16bf v; u16 u[16]; uint4 q[2]; };
union H8     { u16 s[8];  uint4 q; };

// ---- CDNA5 async global->LDS (ASYNCcnt-tracked) ---------------------------
#define LDS_AS __attribute__((address_space(3)))
static __device__ __forceinline__ unsigned lds_off32(const void* p) {
  return (unsigned)(unsigned long long)(LDS_AS const char*)p;   // addrspacecast
}
static __device__ __forceinline__ void async_ld16(unsigned lds_off, const void* g) {
  asm volatile("global_load_async_to_lds_b128 %0, %1, off"
               :: "v"(lds_off), "v"((unsigned long long)g) : "memory");
}
static __device__ __forceinline__ void wait_async0() {
  asm volatile("s_wait_asynccnt 0x0" ::: "memory");
}
static __device__ __forceinline__ void wait_async1() {
  asm volatile("s_wait_asynccnt 0x1" ::: "memory");
}
static __device__ __forceinline__ void wait_async2() {
  asm volatile("s_wait_asynccnt 0x2" ::: "memory");
}

// ---------------------------------------------------------------------------
// GEMM: out[m,n] = (sum_k A[m,k]*W[n,k] + bias[n]) * scale      (x @ W^T + b)
// Block tile 64(M) x 128(N), 8 waves each 32x32 (4 WMMA / K-chunk of 32).
// OUTMODE: 0 = f32 [M,Nc], 1 = bf16 [M,Nc], 2 = bf16 per-head transposed
// [B, H, hd, SEQ] (used for V so attention V tiles are straight copies).
// ---------------------------------------------------------------------------
template<bool A_BF16, int OUTMODE>
__global__ __launch_bounds__(256)
void gemm_nt_bias(const void* __restrict__ A_, const float* __restrict__ W,
                  const float* __restrict__ bias, void* __restrict__ out_,
                  int M, int Nc, int K, float scale) {
  __shared__ u16 As[64 * 40];    // rows padded 32->40 halves: conflict-free frags
  __shared__ u16 Bs[128 * 40];

  const int tid  = threadIdx.x;
  const int lane = tid & 31, wave = tid >> 5;
  const int wm = wave >> 2, wn = wave & 3;          // 2x4 wave grid
  const int halfId = lane >> 4, l15 = lane & 15;
  const int m0 = blockIdx.x * 64, n0 = blockIdx.y * 128;

  v8f cc[2][2];
  #pragma unroll
  for (int i = 0; i < 2; i++)
    #pragma unroll
    for (int j = 0; j < 2; j++) cc[i][j] = v8f{};

  for (int k0 = 0; k0 < K; k0 += 32) {
    __syncthreads();
    { // A tile: 64 rows x 32 k
      int row = tid >> 2, ch = (tid & 3) * 8;
      H8 t;
      if (A_BF16) {
        const u16* A = (const u16*)A_;
        t.q = *(const uint4*)&A[(size_t)(m0 + row) * K + k0 + ch];
      } else {
        const float* A = (const float*)A_;
        const float4* s = (const float4*)&A[(size_t)(m0 + row) * K + k0 + ch];
        float4 x0 = s[0], x1 = s[1];
        t.s[0]=f2bf(x0.x); t.s[1]=f2bf(x0.y); t.s[2]=f2bf(x0.z); t.s[3]=f2bf(x0.w);
        t.s[4]=f2bf(x1.x); t.s[5]=f2bf(x1.y); t.s[6]=f2bf(x1.z); t.s[7]=f2bf(x1.w);
      }
      *(uint4*)&As[row * 40 + ch] = t.q;
    }
    { // B tile: 128 out-cols x 32 k
      int row = tid >> 1, ch = (tid & 1) * 16;
      const float4* s = (const float4*)&W[(size_t)(n0 + row) * K + k0 + ch];
      H8 t0, t1;
      float4 x0 = s[0], x1 = s[1], x2 = s[2], x3 = s[3];
      t0.s[0]=f2bf(x0.x); t0.s[1]=f2bf(x0.y); t0.s[2]=f2bf(x0.z); t0.s[3]=f2bf(x0.w);
      t0.s[4]=f2bf(x1.x); t0.s[5]=f2bf(x1.y); t0.s[6]=f2bf(x1.z); t0.s[7]=f2bf(x1.w);
      t1.s[0]=f2bf(x2.x); t1.s[1]=f2bf(x2.y); t1.s[2]=f2bf(x2.z); t1.s[3]=f2bf(x2.w);
      t1.s[4]=f2bf(x3.x); t1.s[5]=f2bf(x3.y); t1.s[6]=f2bf(x3.z); t1.s[7]=f2bf(x3.w);
      *(uint4*)&Bs[row * 40 + ch]     = t0.q;
      *(uint4*)&Bs[row * 40 + ch + 8] = t1.q;
    }
    __syncthreads();

    const int koffA = halfId ? 16 : 0;    // A frag: K 0-15 / 16-31
    const int koffB = halfId ? 16 : 0;    // B frag: K 0-15 / 16-31
    Frag16 a[2], b[2];
    #pragma unroll
    for (int i = 0; i < 2; i++) {
      const u16* ar = &As[(wm * 32 + i * 16 + l15) * 40];
      a[i].q[0] = *(const uint4*)&ar[koffA];
      a[i].q[1] = *(const uint4*)&ar[8 + koffA];
    }
    #pragma unroll
    for (int j = 0; j < 2; j++) {
      const u16* br = &Bs[(wn * 32 + j * 16 + l15) * 40 + koffB];
      b[j].q[0] = *(const uint4*)&br[0];
      b[j].q[1] = *(const uint4*)&br[8];
    }
    #pragma unroll
    for (int i = 0; i < 2; i++)
      #pragma unroll
      for (int j = 0; j < 2; j++)
        cc[i][j] = wmma_bf16(a[i].v, b[j].v, cc[i][j]);
  }

  #pragma unroll
  for (int i = 0; i < 2; i++)
    #pragma unroll
    for (int j = 0; j < 2; j++) {
      int col = n0 + wn * 32 + j * 16 + l15;
      float bv = bias[col];
      if (OUTMODE == 2) {
        // token0..token0+7 are consecutive and stay inside one batch
        int token0 = m0 + wm * 32 + i * 16 + 8 * halfId;
        int bidx = token0 >> 11, n = token0 & (SEQ - 1);
        int hh = col >> 6, d = col & (HD - 1);
        H8 t;
        #pragma unroll
        for (int r = 0; r < 8; r++) t.s[r] = f2bf((cc[i][j][r] + bv) * scale);
        *(uint4*)&((u16*)out_)[(((size_t)(bidx * NH + hh) * HD + d) * SEQ) + n] = t.q;
      } else {
        #pragma unroll
        for (int r = 0; r < 8; r++) {
          int row = m0 + wm * 32 + i * 16 + r + 8 * halfId;
          float val = (cc[i][j][r] + bv) * scale;
          if (OUTMODE == 1) ((u16*)out_)[(size_t)row * Nc + col] = f2bf(val);
          else              ((float*)out_)[(size_t)row * Nc + col] = val;
        }
      }
    }
}

// ---------------------------------------------------------------------------
// Streaming attention. grid = (SEQ/128, BATCH*NH), 256 threads (8 waves).
// Wave handles 16 query rows. Pass 1: lane-local online max/sumexp (zero
// cross-lane traffic in the loop), one (m,Z) butterfly merge at the end.
// Pass 2: p = exp(s-m)/Z, ctx += p @ v (WMMA), per-key column sums.
// K/V tiles: async global->LDS, double-buffered. V comes pre-transposed.
// ---------------------------------------------------------------------------
__global__ __launch_bounds__(256)
void attn_kernel(const u16* __restrict__ qb, const u16* __restrict__ kb,
                 const u16* __restrict__ vbT, u16* __restrict__ ctxb,
                 float* __restrict__ mean_attn) {
  __shared__ u16 Ks[2][32 * 72];    // [key][hd], rows padded 64->72
  __shared__ u16 Vs[2][64 * 40];    // [hd col][key], rows padded 32->40
  __shared__ u16 Ps[8 * 16 * 40];   // per-wave p staging
  __shared__ float colsum[SEQ];

  const int tid  = threadIdx.x;
  const int lane = tid & 31, wave = tid >> 5;
  const int halfId = lane >> 4, l15 = lane & 15;
  const int b = blockIdx.y / NH, h = blockIdx.y % NH;
  const int q0 = blockIdx.x * 128 + wave * 16;
  const int NT = SEQ / 32;

  for (int i = tid; i < SEQ; i += 256) colsum[i] = 0.f;

  const int krow = tid >> 3, kch = (tid & 7) * 8;   // K tile: 32 x 64
  const int vrow = tid >> 2, vch = (tid & 3) * 8;   // V tile: 64 x 32
  const size_t kbase = (size_t)(b * SEQ) * EMB + h * HD;
  const size_t vbase = (size_t)(b * NH + h) * HD * SEQ;

  auto issueK = [&](int tile, int buf) {
    async_ld16(lds_off32(&Ks[buf][krow * 72 + kch]),
               &kb[kbase + (size_t)(tile * 32 + krow) * EMB + kch]);
  };
  auto issueV = [&](int tile, int buf) {
    async_ld16(lds_off32(&Vs[buf][vrow * 40 + vch]),
               &vbT[vbase + (size_t)vrow * SEQ + tile * 32 + vch]);
  };

  // q fragments for K-dim chunks 0-31 and 32-63 (held for whole kernel)
  Frag16 aq0, aq1;
  {
    const u16* qr = &qb[((size_t)(b * SEQ + q0 + l15) * EMB) + h * HD];
    int ko = halfId ? 16 : 0;
    aq0.q[0] = *(const uint4*)&qr[ko];
    aq0.q[1] = *(const uint4*)&qr[8 + ko];
    aq1.q[0] = *(const uint4*)&qr[32 + ko];
    aq1.q[1] = *(const uint4*)&qr[40 + ko];
  }

  const int koffB = halfId ? 16 : 0;
  float ml[8], zp[8];                     // lane-local running max / partial sum
  #pragma unroll
  for (int r = 0; r < 8; r++) { ml[r] = -1e30f; zp[r] = 0.f; }

  // ---------------- pass 1: lane-local online max + sumexp ----------------
  issueK(0, 0);
  for (int it = 0; it < NT; ++it) {
    int cur = it & 1;
    __syncthreads();                      // prior consumers of buf[cur^1] done
    bool more = (it + 1 < NT);
    if (more) { issueK(it + 1, cur ^ 1); wait_async1(); } else { wait_async0(); }
    __syncthreads();                      // all waves' tile-it data in LDS

    #pragma unroll
    for (int t = 0; t < 2; t++) {
      Frag16 bk0, bk1;
      const u16* kr = &Ks[cur][(t * 16 + l15) * 72 + koffB];
      bk0.q[0] = *(const uint4*)&kr[0];
      bk0.q[1] = *(const uint4*)&kr[8];
      bk1.q[0] = *(const uint4*)&kr[32];
      bk1.q[1] = *(const uint4*)&kr[40];
      v8f s = v8f{};
      s = wmma_bf16(aq0.v, bk0.v, s);
      s = wmma_bf16(aq1.v, bk1.v, s);
      #pragma unroll
      for (int r = 0; r < 8; r++) {       // per-lane, no shuffles
        float v = s[r];
        float mn = fmaxf(ml[r], v);
        zp[r] = zp[r] * __expf(ml[r] - mn) + __expf(v - mn);
        ml[r] = mn;
      }
    }
  }

  // single cross-lane (m,Z) merge: butterfly within each 16-lane half
  float mrun[8], invz[8];
  #pragma unroll
  for (int r = 0; r < 8; r++) {
    float m = ml[r], z = zp[r];
    #pragma unroll
    for (int off = 1; off < 16; off <<= 1) {
      float mo = __shfl_xor(m, off, 32);
      float zo = __shfl_xor(z, off, 32);
      float mn = fmaxf(m, mo);
      z = z * __expf(m - mn) + zo * __expf(mo - mn);
      m = mn;
    }
    mrun[r] = m;
    invz[r] = 1.0f / z;
  }

  // ---------------- pass 2: p, ctx += p @ v, column sums ----------------
  v8f ctx[4];
  #pragma unroll
  for (int ct = 0; ct < 4; ct++) ctx[ct] = v8f{};

  issueK(0, 0); issueV(0, 0);
  for (int it = 0; it < NT; ++it) {
    int cur = it & 1;
    int kc = it * 32;
    __syncthreads();
    bool more = (it + 1 < NT);
    if (more) { issueK(it + 1, cur ^ 1); issueV(it + 1, cur ^ 1); wait_async2(); }
    else      { wait_async0(); }
    __syncthreads();

    u16* pw = &Ps[wave * 640];
    #pragma unroll
    for (int t = 0; t < 2; t++) {
      Frag16 bk0, bk1;
      const u16* kr = &Ks[cur][(t * 16 + l15) * 72 + koffB];
      bk0.q[0] = *(const uint4*)&kr[0];
      bk0.q[1] = *(const uint4*)&kr[8];
      bk1.q[0] = *(const uint4*)&kr[32];
      bk1.q[1] = *(const uint4*)&kr[40];
      v8f s = v8f{};
      s = wmma_bf16(aq0.v, bk0.v, s);
      s = wmma_bf16(aq1.v, bk1.v, s);
      float cs = 0.f;
      #pragma unroll
      for (int r = 0; r < 8; r++) {
        float p = __expf(s[r] - mrun[r]) * invz[r];   // final probability
        s[r] = p;
        cs += p;
      }
      cs += __shfl_xor(cs, 16, 32);                   // combine row halves
      if (lane < 16) atomicAdd(&colsum[kc + t * 16 + lane], cs);
      #pragma unroll
      for (int r = 0; r < 8; r++)                     // C-layout -> staging
        pw[(r + 8 * halfId) * 40 + t * 16 + l15] = f2bf(s[r]);
    }

    // p back in A-fragment layout (wave-local; DS ops in-order)
    Frag16 ap;
    const u16* pr = &pw[l15 * 40 + (halfId ? 16 : 0)];
    ap.q[0] = *(const uint4*)&pr[0];
    ap.q[1] = *(const uint4*)&pr[8];

    #pragma unroll
    for (int ct = 0; ct < 4; ct++) {
      Frag16 bv;
      const u16* vr = &Vs[cur][(ct * 16 + l15) * 40 + koffB];
      bv.q[0] = *(const uint4*)&vr[0];
      bv.q[1] = *(const uint4*)&vr[8];
      ctx[ct] = wmma_bf16(ap.v, bv.v, ctx[ct]);
    }
  }

  // write ctx (already normalized) as bf16, layout [B,N,D]
  #pragma unroll
  for (int ct = 0; ct < 4; ct++)
    #pragma unroll
    for (int r = 0; r < 8; r++) {
      int row = q0 + r + 8 * halfId;
      int col = h * HD + ct * 16 + l15;
      ctxb[(size_t)(b * SEQ + row) * EMB + col] = f2bf(ctx[ct][r]);
    }

  __syncthreads();
  for (int i = tid; i < SEQ; i += 256)
    atomicAdd(&mean_attn[b * SEQ + i], colsum[i]);
}

// ---------------------------------------------------------------------------
// Per-batch bitonic sort of (mean_attn, index) descending (tie: lower index
// first, matching jax.lax.top_k), then gather top_k value rows.
// ---------------------------------------------------------------------------
__global__ __launch_bounds__(1024)
void topk_gather(const float* __restrict__ mean_attn,
                 const float* __restrict__ value,
                 const int* __restrict__ topk_ptr,
                 float* __restrict__ sel_out) {
  __shared__ float sv[SEQ];
  __shared__ int   si[SEQ];
  const int b = blockIdx.x, tid = threadIdx.x;
  for (int i = tid; i < SEQ; i += 1024) { sv[i] = mean_attn[b * SEQ + i]; si[i] = i; }
  __syncthreads();
  for (int k = 2; k <= SEQ; k <<= 1)
    for (int j = k >> 1; j > 0; j >>= 1) {
      for (int i = tid; i < SEQ; i += 1024) {
        int ixj = i ^ j;
        if (ixj > i) {
          float a = sv[i], c = sv[ixj];
          int  ia = si[i], ic = si[ixj];
          bool a_first = (a > c) || (a == c && ia < ic);
          bool desc = ((i & k) == 0);
          if (desc ? !a_first : a_first) {
            sv[i] = c; sv[ixj] = a; si[i] = ic; si[ixj] = ia;
          }
        }
      }
      __syncthreads();
    }
  const int TK = *topk_ptr;
  for (int e = tid; e < TK * EMB; e += 1024) {
    int i = e / EMB, d = e - i * EMB;
    sel_out[(size_t)b * TK * EMB + e] =
        value[((size_t)b * SEQ + si[i]) * EMB + d];
  }
}

// ---------------------------------------------------------------------------
extern "C" void kernel_launch(void* const* d_in, const int* in_sizes, int n_in,
                              void* d_out, int out_size, void* d_ws, size_t ws_size,
                              hipStream_t stream) {
  (void)in_sizes; (void)n_in; (void)out_size; (void)ws_size;
  const float* query = (const float*)d_in[0];
  const float* key   = (const float*)d_in[1];
  const float* value = (const float*)d_in[2];
  const float* Wq = (const float*)d_in[3];  const float* bq = (const float*)d_in[4];
  const float* Wk = (const float*)d_in[5];  const float* bk = (const float*)d_in[6];
  const float* Wv = (const float*)d_in[7];  const float* bv = (const float*)d_in[8];
  const float* Wo = (const float*)d_in[9];  const float* bo = (const float*)d_in[10];
  const int*   topk = (const int*)d_in[11];

  const int M = BATCH * SEQ;                       // 8192
  float* out = (float*)d_out;                      // [B,N,D]
  float* sel = out + (size_t)M * EMB;              // [B,top_k,D]

  const size_t SZ = (size_t)M * EMB;               // elements per bf16 tensor
  u16* qb   = (u16*)d_ws;
  u16* kb   = qb + SZ;
  u16* vbT  = kb + SZ;                             // [B,H,hd,SEQ]
  u16* ctxb = vbT + SZ;
  float* mean_attn = (float*)(ctxb + SZ);          // [B,SEQ]

  dim3 gG(M / 64, EMB / 128);                      // 128 x 6
  const float qscale = 0.125f;                     // 1/sqrt(HD)

  gemm_nt_bias<false, 1><<<gG, 256, 0, stream>>>(query, Wq, bq, qb, M, EMB, EMB, qscale);
  gemm_nt_bias<false, 1><<<gG, 256, 0, stream>>>(key,   Wk, bk, kb, M, EMB, EMB, 1.0f);
  gemm_nt_bias<false, 2><<<gG, 256, 0, stream>>>(value, Wv, bv, vbT, M, EMB, EMB, 1.0f);

  hipMemsetAsync(mean_attn, 0, (size_t)BATCH * SEQ * sizeof(float), stream);

  attn_kernel<<<dim3(SEQ / 128, BATCH * NH), 256, 0, stream>>>(qb, kb, vbT, ctxb, mean_attn);

  gemm_nt_bias<true, 0><<<gG, 256, 0, stream>>>(ctxb, Wo, bo, out, M, EMB, EMB, 1.0f);

  topk_gather<<<BATCH, 1024, 0, stream>>>(mean_attn, value, topk, sel);
}
